// conv_attention_66228395704953
// MI455X (gfx1250) — matrix operand order repaired and checked by reading, not compiled, not run on hardware
//
#include <hip/hip_runtime.h>

// Problem constants
#define BB 16
#define HH 8
#define TT 512
#define DD 64
#define NHEADS (BB*HH)          // 128
#define TILE 32                 // rows of t per block
#define NT (TT/TILE)            // 16 tiles
#define LDA_A 520               // bf16 row stride (4-bank skew vs 512)

typedef __attribute__((ext_vector_type(16))) __bf16 v16bf;
typedef __attribute__((ext_vector_type(8)))  float  v8f;

union Frag {
    v16bf v;
    uint4 q[2];
};

__device__ __forceinline__ unsigned short f2bf(float f) {
    unsigned int u = __float_as_uint(f);
    unsigned int r = u + 0x7FFFu + ((u >> 16) & 1u);   // round-to-nearest-even
    return (unsigned short)(r >> 16);
}

__device__ __forceinline__ v8f wmma_bf16(const Frag& a, const Frag& b, v8f c) {
    return __builtin_amdgcn_wmma_f32_16x16x32_bf16(
        /*neg_a=*/false, a.v, /*neg_b=*/false, b.v,
        /*c_mod=*/(short)0, c, /*reuse_a=*/false, /*reuse_b=*/false);
}

// ---------------- precompute kernels ----------------

__global__ void k_cvt_bf16(const float* __restrict__ in,
                           unsigned short* __restrict__ out, int n) {
    int i = blockIdx.x * 256 + threadIdx.x;
    if (i < n) out[i] = f2bf(in[i]);
}

// V pre-swizzled into WMMA B-fragment order:
// out[((n*16+ic)*4+dt)*512 + lane*16 + j] = V[n][ic*32 + (lane>>4)*16 + j][dt*16 + lane%16]
__global__ void k_vfrag(const float* __restrict__ V, unsigned short* __restrict__ out) {
    int tid = blockIdx.x * 256 + threadIdx.x;           // 4,194,304 total
    int j = tid & 15, l = (tid >> 4) & 31, tile = tid >> 9;
    int dt = tile & 3, ic = (tile >> 2) & 15, n = tile >> 6;
    int s = ic * 32 + ((l >> 4) << 4) + j;
    int d = dt * 16 + (l & 15);
    out[tid] = f2bf(V[(size_t)n * (TT * DD) + s * DD + d]);
}

// W^T pre-swizzled B-fragments per (k, i-chunk, o-tile):
// out[((k*16+ic)*32+ct)*512 + lane*16 + j] = conv_w[o][i][k],
//   o = ct*16 + lane%16, i = ic*32 + (lane>>4)*16 + j
__global__ void k_wtfrag(const float* __restrict__ W, unsigned short* __restrict__ out) {
    int tid = blockIdx.x * 256 + threadIdx.x;           // 786,432 total
    int j = tid & 15, l = (tid >> 4) & 31, tile = tid >> 9;
    int ct = tile & 31, ic = (tile >> 5) & 15, k = tile >> 9;
    int o = ct * 16 + (l & 15);
    int i = ic * 32 + ((l >> 4) << 4) + j;
    out[tid] = f2bf(W[(size_t)(o * TT + i) * 3 + k]);
}

// ---------------- fused attention+conv kernel ----------------

__global__ __launch_bounds__(256)
void k_fused(const unsigned short* __restrict__ Qbf,
             const unsigned short* __restrict__ Kbf,
             const unsigned short* __restrict__ Vfrag,
             const unsigned short* __restrict__ WTfrag,
             const int* __restrict__ mask,
             const float* __restrict__ conv_b,
             float* __restrict__ outp,
             float* __restrict__ p_attn) {
    __shared__ unsigned short ldsA[(TILE + 2) * LDA_A];  // 35,360 B (scores in, later P bf16)
    __shared__ float redM[TILE * 4];                     // per-row, per-colgroup max partials
    __shared__ float redS[TILE * 4];                     // per-row, per-colgroup sum partials

    const int tid  = threadIdx.x;
    const int wave = tid >> 5;
    const int lane = tid & 31;
    const int h    = lane >> 4;     // half-wave
    const int nloc = lane & 15;     // N / row-in-strip index
    const int n    = blockIdx.x >> 4;           // head index 0..127
    const int t0   = (blockIdx.x & 15) * TILE;  // row tile origin
    const int bidx = n >> 3;                    // batch for mask

    // ---------- Stage 0: masked scores (34 rows incl. halo) -> ldsA (bf16) ----------
    // 3 row-strips x 32 col-tiles = 96 WMMA tiles round-robined over 8 waves
    for (int idx = wave; idx < 96; idx += 8) {
        const int strip = idx >> 5;
        const int ct    = idx & 31;
        v8f acc = {0.f, 0.f, 0.f, 0.f, 0.f, 0.f, 0.f, 0.f};

        int rowt = t0 - 1 + strip * 16 + nloc;               // A-matrix row (M = lane%16)
        int rowc = rowt < 0 ? 0 : (rowt > (TT - 1) ? (TT - 1) : rowt);
        const unsigned short* qrow = Qbf + (size_t)rowc * DD;
        const unsigned short* krow = Kbf + (size_t)n * (TT * DD) + (ct * 16 + nloc) * DD;

        #pragma unroll
        for (int ic = 0; ic < 2; ++ic) {                     // K-dim D=64 in 2 chunks of 32
            Frag a, b;
            int c0 = ic * 32 + (h << 3);
            a.q[0] = *(const uint4*)(qrow + c0);
            a.q[1] = *(const uint4*)(qrow + c0 + 16);
            const unsigned short* kb = krow + ic * 32 + (h << 3);
            b.q[0] = *(const uint4*)(kb);
            b.q[1] = *(const uint4*)(kb + 16);
            acc = wmma_bf16(a, b, acc);
        }

        // Branch-free mask epilogue: 8 independent clamped mask loads (one clause),
        // then pure v_cndmask selects; only the LDS store is predicated.
        const int s = ct * 16 + nloc;                        // D-matrix N = lane%16
        const int* mcol = mask + (size_t)bidx * (TT * TT) + s;
        int mv[8];
        #pragma unroll
        for (int r = 0; r < 8; ++r) {
            int lrow = strip * 16 + r + (h << 3);            // D-matrix M = r + 8*half
            int t = t0 - 1 + lrow;
            int tc = t < 0 ? 0 : (t > (TT - 1) ? (TT - 1) : t);
            mv[r] = mcol[tc * TT];                           // independent -> s_clause
        }
        #pragma unroll
        for (int r = 0; r < 8; ++r) {
            int lrow = strip * 16 + r + (h << 3);
            int t = t0 - 1 + lrow;
            bool inr = (t >= 0) && (t < TT);                 // halo zero-padding
            float val = (mv[r] != 0) ? acc[r] * 0.125f : -1e9f;
            unsigned short vb = inr ? f2bf(val) : (unsigned short)0;
            if (lrow < TILE + 2)
                ldsA[lrow * LDA_A + s] = vb;
        }
    }
    __syncthreads();

    // ---------- Stage 1 + 2: conv matmul, then register-resident softmax ----------
    unsigned short* ldsP = ldsA;           // reuse region (barrier-separated)
    {
        const int strip  = wave & 1;            // 2 row-strips
        const int wgrp   = wave >> 1;           // 4 column groups
        const int ctbase = wgrp << 3;           // 8 col-tiles per wave (of 32)
        v8f acc[8];
        v8f zero = {0.f, 0.f, 0.f, 0.f, 0.f, 0.f, 0.f, 0.f};
        #pragma unroll
        for (int i = 0; i < 8; ++i) acc[i] = zero;

        // S2 = sum_k shift_k(S) @ Wk^T
        for (int k = 0; k < 3; ++k) {
            #pragma unroll 2
            for (int ic = 0; ic < 16; ++ic) {
                // Prefetch next i-chunk's B fragments (lanes span the 8KB region)
                {
                    int icn = ic + 1;
                    int kn  = k;
                    if (icn == 16) { icn = 0; kn = k + 1; }
                    if (kn < 3) {
                        const unsigned short* pf = WTfrag
                            + (size_t)((kn * 16 + icn) * 32 + ctbase) * 512
                            + lane * 128;
                        __builtin_prefetch(pf, 0, 3);        // global_prefetch_b8
                    }
                }
                Frag a;
                int arow = strip * 16 + nloc + k;            // input row = t + k - 1 (halo offset)
                int c0 = ic * 32 + (h << 4);
                const unsigned short* ap = &ldsA[arow * LDA_A + c0];
                a.q[0] = *(const uint4*)ap;
                a.q[1] = *(const uint4*)(ap + 8);
                const unsigned short* wb = WTfrag + (size_t)((k * 16 + ic) * 32) * 512;
                #pragma unroll
                for (int ct = 0; ct < 8; ++ct) {
                    const unsigned short* bp = wb + (ctbase + ct) * 512 + lane * 16;
                    Frag b;
                    b.q[0] = *(const uint4*)bp;
                    b.q[1] = *(const uint4*)(bp + 8);
                    acc[ct] = wmma_bf16(a, b, acc[ct]);
                }
            }
        }
        // Fold conv bias into accumulators
        #pragma unroll
        for (int ct = 0; ct < 8; ++ct) {
            float bv = conv_b[(ctbase + ct) * 16 + nloc];
            #pragma unroll
            for (int r = 0; r < 8; ++r) acc[ct][r] += bv;
        }

        // Row max: in-register over 8 col-tiles, shuffle over the 16 lanes of a row,
        // cross-wave via 512B LDS partials.
        #pragma unroll
        for (int r = 0; r < 8; ++r) {
            float m = acc[0][r];
            #pragma unroll
            for (int ct = 1; ct < 8; ++ct) m = fmaxf(m, acc[ct][r]);
            #pragma unroll
            for (int x = 1; x < 16; x <<= 1) m = fmaxf(m, __shfl_xor(m, x, 32));
            if (nloc == 0) redM[(strip * 16 + r + (h << 3)) * 4 + wgrp] = m;
        }
        __syncthreads();

        // exp in place + row sum (same reduction tree)
        #pragma unroll
        for (int r = 0; r < 8; ++r) {
            int rho = strip * 16 + r + (h << 3);
            float m0 = fmaxf(fmaxf(redM[rho * 4 + 0], redM[rho * 4 + 1]),
                             fmaxf(redM[rho * 4 + 2], redM[rho * 4 + 3]));
            float ssum = 0.f;
            #pragma unroll
            for (int ct = 0; ct < 8; ++ct) {
                float e = __expf(acc[ct][r] - m0);
                acc[ct][r] = e;
                ssum += e;
            }
            #pragma unroll
            for (int x = 1; x < 16; x <<= 1) ssum += __shfl_xor(ssum, x, 32);
            if (nloc == 0) redS[rho * 4 + wgrp] = ssum;
        }
        __syncthreads();

        // Normalize: write p_attn fp32 to global and bf16 P into LDS for P@V
        float* pbase = p_attn + (size_t)n * (TT * TT) + (size_t)t0 * TT;
        #pragma unroll
        for (int r = 0; r < 8; ++r) {
            int rho = strip * 16 + r + (h << 3);
            float inv = 1.0f / (redS[rho * 4 + 0] + redS[rho * 4 + 1] +
                                redS[rho * 4 + 2] + redS[rho * 4 + 3]);
            #pragma unroll
            for (int ct = 0; ct < 8; ++ct) {
                int col = (ctbase + ct) * 16 + nloc;
                float p = acc[ct][r] * inv;
                pbase[(size_t)rho * TT + col] = p;
                ldsP[rho * LDA_A + col] = f2bf(p);
            }
        }
    }
    __syncthreads();

    // ---------- Stage 3: out = P @ V (reduce over o=512 in 16 chunks) ----------
    {
        const int strip = wave & 1;
        const int dt    = wave >> 1;    // 4 d-tiles
        v8f acc = {0.f, 0.f, 0.f, 0.f, 0.f, 0.f, 0.f, 0.f};
        #pragma unroll 2
        for (int ic = 0; ic < 16; ++ic) {
            Frag a, b;
            int arow = strip * 16 + nloc;
            int c0 = ic * 32 + (h << 4);
            const unsigned short* ap = &ldsP[arow * LDA_A + c0];
            a.q[0] = *(const uint4*)ap;
            a.q[1] = *(const uint4*)(ap + 8);
            const unsigned short* bp =
                Vfrag + (size_t)(((n * 16 + ic) * 4 + dt)) * 512 + lane * 16;
            b.q[0] = *(const uint4*)bp;
            b.q[1] = *(const uint4*)(bp + 8);
            acc = wmma_bf16(a, b, acc);
        }
        int d = dt * 16 + nloc;
        #pragma unroll
        for (int r = 0; r < 8; ++r) {
            int t = t0 + strip * 16 + r + (h << 3);
            outp[(size_t)n * (TT * DD) + t * DD + d] = acc[r];
        }
    }
}

// ---------------- launcher ----------------

extern "C" void kernel_launch(void* const* d_in, const int* in_sizes, int n_in,
                              void* d_out, int out_size, void* d_ws, size_t ws_size,
                              hipStream_t stream) {
    (void)in_sizes; (void)n_in; (void)out_size; (void)ws_size;
    const float* query  = (const float*)d_in[0];
    const float* key    = (const float*)d_in[1];
    const float* value  = (const float*)d_in[2];
    const int*   mask   = (const int*)d_in[3];
    const float* conv_w = (const float*)d_in[4];
    const float* conv_b = (const float*)d_in[5];

    float* outp   = (float*)d_out;
    float* p_attn = outp + (size_t)NHEADS * TT * DD;       // 4,194,304 floats

    char* ws = (char*)d_ws;
    unsigned short* Qbf    = (unsigned short*)(ws);                 // 8 MB
    unsigned short* Kbf    = (unsigned short*)(ws + 8388608);       // 8 MB
    unsigned short* Vfrag  = (unsigned short*)(ws + 16777216);      // 8 MB
    unsigned short* WTfrag = (unsigned short*)(ws + 25165824);      // 1.5 MB

    const int nQK = NHEADS * TT * DD;                       // 4,194,304
    k_cvt_bf16<<<nQK / 256, 256, 0, stream>>>(query, Qbf, nQK);
    k_cvt_bf16<<<nQK / 256, 256, 0, stream>>>(key,   Kbf, nQK);
    k_vfrag  <<<nQK / 256, 256, 0, stream>>>(value, Vfrag);
    k_wtfrag <<<(3 * TT * TT) / 256, 256, 0, stream>>>(conv_w, WTfrag);

    k_fused<<<NHEADS * NT, 256, 0, stream>>>(Qbf, Kbf, Vfrag, WTfrag,
                                             mask, conv_b, outp, p_attn);
}
